// VoxAttention_49349174231258
// MI455X (gfx1250) — matrix operand order repaired and checked by reading, not compiled, not run on hardware
//
#include <hip/hip_runtime.h>
#include <stdint.h>

// ---------------------------------------------------------------------------
// VoxAttention for MI455X (gfx1250, wave32, WMMA).
// B=8, W=H=32 (N=1024 tokens), C=64, NH=8, P=32, scale=sqrt(64)=8.
// scores(b,i,j,h,k,l) = (QK^T + q_row.row_emb[k-i+31] + q_col.col_emb[l-j+31])/8
// probs = softmax over (k,l); mixed = probs @ hidden_states; out = mixed @ Wv^T + bv
//
// Workspace layout (requires ws_size >= 49 MB):
//   Qbf  [b][h][n][d]  bf16   8 MB   @ 0
//   Kbf  [b][h][n][d]  bf16   8 MB   @ 8 MB
//   Vbf  [b][d][n]     bf16   1 MB   @ 16 MB   (d-major => V tiles pre-transposed)
//   RS   [b][h][n][k]  f32    8 MB   @ 17 MB
//   CS   [b][h][n][l]  f32    8 MB   @ 25 MB
//   OB   [b][n][h*64+d] f32  16 MB   @ 33 MB
// ---------------------------------------------------------------------------

typedef __attribute__((ext_vector_type(16))) __bf16    v16bf;
typedef __attribute__((ext_vector_type(8)))  float     v8f;
typedef __attribute__((ext_vector_type(8)))  uint32_t  v8u;

__device__ __forceinline__ uint16_t f2bf_bits(float f) {
  uint32_t u = __float_as_uint(f);
  u += 0x7FFFu + ((u >> 16) & 1u);      // round-to-nearest-even
  return (uint16_t)(u >> 16);
}
__device__ __forceinline__ float bf2f(uint16_t b) {
  return __uint_as_float(((uint32_t)b) << 16);
}

// ---------------------------------------------------------------------------
// Kernel 1: Q/K projections (+bias) -> bf16, and hidden_states -> bf16 d-major V.
// One block per token (B*N = 8192 blocks).
// ---------------------------------------------------------------------------
__global__ __launch_bounds__(256)
void vox_proj(const float* __restrict__ hs,
              const float* __restrict__ Wq, const float* __restrict__ bq,
              const float* __restrict__ Wk, const float* __restrict__ bk,
              uint16_t* __restrict__ Qbf, uint16_t* __restrict__ Kbf,
              uint16_t* __restrict__ Vbf) {
  __shared__ float row[64];
  const int bn = blockIdx.x;
  const int b = bn >> 10, n = bn & 1023;
  const int t = threadIdx.x;
  if (t < 64) row[t] = hs[(size_t)bn * 64 + t];
  __syncthreads();
  if (t < 64) Vbf[((size_t)b * 64 + t) * 1024 + n] = f2bf_bits(row[t]);
#pragma unroll
  for (int rep = 0; rep < 2; ++rep) {
    const int o = t + rep * 256;                  // output channel 0..511
    const float* wq = Wq + (size_t)o * 64;
    const float* wk = Wk + (size_t)o * 64;
    float aq = bq[o], ak = bk[o];
#pragma unroll
    for (int c = 0; c < 64; ++c) { aq += row[c] * wq[c]; ak += row[c] * wk[c]; }
    const int h = o >> 6, d = o & 63;
    const size_t dst = (((size_t)b * 8 + h) * 1024 + n) * 64 + d;
    Qbf[dst] = f2bf_bits(aq);
    Kbf[dst] = f2bf_bits(ak);
  }
}

// ---------------------------------------------------------------------------
// Kernel 2: separable relative-position scores.
// RS[b,h,n,k] = sum_d q[b,n,h,d]      * row_emb[k-i+31, d]   (d<32)
// CS[b,h,n,l] = sum_d q[b,n,h,32+d]   * col_emb[l-j+31, d]
// One block per token; 256 threads = 8 heads x 32 (k or l) indices.
// ---------------------------------------------------------------------------
__global__ __launch_bounds__(256)
void vox_bias(const uint16_t* __restrict__ Qbf,
              const float* __restrict__ row_emb, const float* __restrict__ col_emb,
              float* __restrict__ RS, float* __restrict__ CS) {
  const int bn = blockIdx.x;
  const int b = bn >> 10, n = bn & 1023;
  const int i = n >> 5, j = n & 31;
  const int t = threadIdx.x & 31, h = threadIdx.x >> 5;
  const uint16_t* q = Qbf + (((size_t)b * 8 + h) * 1024 + n) * 64;
  const float* re = row_emb + (size_t)(t - i + 31) * 32;
  const float* ce = col_emb + (size_t)(t - j + 31) * 32;
  float rs = 0.f, cs = 0.f;
#pragma unroll
  for (int d = 0; d < 32; ++d) {
    rs += bf2f(q[d])      * re[d];
    cs += bf2f(q[32 + d]) * ce[d];
  }
  const size_t base = (((size_t)b * 8 + h) * 1024 + n) * 32 + t;
  RS[base] = rs;
  CS[base] = cs;
}

// ---------------------------------------------------------------------------
// Kernel 3: flash attention, one wave32 per (b,h, 16-query tile).
// Key tiles of 32 keys (= one image row k), 32 tiles total.
// WMMA fragment layouts per CDNA5 ISA 7.12.2:
//   A (16x32 bf16): lane L -> row M=L&15; VGPR r holds K pair
//       k0 = 2r + 16*(L>>4) (contiguous order used here, not the ISA table)
//   B (32x16 bf16): lane L -> col N=L&15; VGPR r holds K pair k0 = 2r + 16*(L>>4) in this code, same order as A
//   C/D (16x16 f32): VGPR v -> row M = v + 8*(L>>4), col N = L&15
// ---------------------------------------------------------------------------
#define KT_STRIDE 72   // halfs; 144B rows: 16B aligned, conflict-free gathers
#define VT_STRIDE 40   // halfs; 80B rows
#define PT_STRIDE 34   // halfs; 68B rows

__global__ __launch_bounds__(32)
void vox_attn(const uint16_t* __restrict__ Qbf, const uint16_t* __restrict__ Kbf,
              const uint16_t* __restrict__ Vbf, const float* __restrict__ RS,
              const float* __restrict__ CS, float* __restrict__ OB) {
  __shared__ uint16_t Kt[32 * KT_STRIDE];   // K tile (also Q staging)
  __shared__ uint16_t Vt[64 * VT_STRIDE];   // V tile, [d][key]
  __shared__ uint16_t Pt[16 * PT_STRIDE];   // probs tile, [q][key]
  __shared__ float RSt[512], CSt[512];      // 16 queries x 32

  const int bh = blockIdx.y;                // b*8 + h
  const int b  = bh >> 3;
  const int n0 = blockIdx.x * 16;           // query tile base (fixed image row i)
  const int lane = threadIdx.x;
  const int lh   = lane >> 4;               // half-wave
  const int col  = lane & 15;

  const uint16_t* Qg = Qbf + (((size_t)bh) * 1024 + n0) * 64;
  const uint16_t* Kg = Kbf + ((size_t)bh) * 1024 * 64;
  const uint16_t* Vg = Vbf + ((size_t)b) * 64 * 1024;

  // ---- stage Q (16x64) into Kt, RS/CS rows into LDS ----
#pragma unroll
  for (int it = 0; it < 4; ++it) {
    const int idx = lane + 32 * it, r = idx >> 3, s = idx & 7;
    *(uint4*)&Kt[r * KT_STRIDE + s * 8] = *(const uint4*)&Qg[r * 64 + s * 8];
  }
  {
    const float* rsg = RS + (((size_t)bh) * 1024 + n0) * 32;
    const float* csg = CS + (((size_t)bh) * 1024 + n0) * 32;
    for (int idx = lane; idx < 512; idx += 32) { RSt[idx] = rsg[idx]; CSt[idx] = csg[idx]; }
  }
  __syncthreads();

  // ---- gather Q A-fragments (d slices 0-31, 32-63) ----
  v8u aqu0, aqu1;
#pragma unroll
  for (int r = 0; r < 8; ++r) {
    const int k = 2 * r + 16 * lh;
    aqu0[r] = *(const uint32_t*)&Kt[col * KT_STRIDE + 0  + k];
    aqu1[r] = *(const uint32_t*)&Kt[col * KT_STRIDE + 32 + k];
  }
  const v16bf aq0 = __builtin_bit_cast(v16bf, aqu0);
  const v16bf aq1 = __builtin_bit_cast(v16bf, aqu1);
  __syncthreads();

  v8f o0 = {}, o1 = {}, o2 = {}, o3 = {};
  float rowm[8], rowl[8];
#pragma unroll
  for (int v = 0; v < 8; ++v) { rowm[v] = -1e30f; rowl[v] = 0.f; }

  for (int kt = 0; kt < 32; ++kt) {
    const int m0 = kt * 32;
    // ---- stage K tile (32 keys x 64 d) and V tile ([d][key], pre-transposed) ----
#pragma unroll
    for (int it = 0; it < 8; ++it) {
      const int idx = lane + 32 * it, r = idx >> 3, s = idx & 7;
      *(uint4*)&Kt[r * KT_STRIDE + s * 8] = *(const uint4*)&Kg[(size_t)(m0 + r) * 64 + s * 8];
    }
#pragma unroll
    for (int it = 0; it < 8; ++it) {
      const int idx = lane + 32 * it, d = idx >> 2, s = idx & 3;
      *(uint4*)&Vt[d * VT_STRIDE + s * 8] = *(const uint4*)&Vg[(size_t)d * 1024 + m0 + s * 8];
    }
    if (kt + 1 < 32) {                                   // global_prefetch_b8
      __builtin_prefetch(&Kg[(size_t)(m0 + 32 + lane) * 64], 0, 0);
      __builtin_prefetch(&Vg[(size_t)(lane * 2) * 1024 + m0 + 32], 0, 0);
    }
    __syncthreads();

    // ---- S = Q * K^T : two 16x16 f32 blocks (keys 0-15, 16-31 of tile) ----
    v8f s0 = {}, s1 = {};
    {
      v8u bu; v16bf bf;
#pragma unroll
      for (int r = 0; r < 8; ++r) bu[r] = *(const uint32_t*)&Kt[col * KT_STRIDE + 0 + (2 * r + 16 * lh)];
      bf = __builtin_bit_cast(v16bf, bu);
      s0 = __builtin_amdgcn_wmma_f32_16x16x32_bf16(false, aq0, false, bf, (short)0, s0, false, false);
#pragma unroll
      for (int r = 0; r < 8; ++r) bu[r] = *(const uint32_t*)&Kt[col * KT_STRIDE + 32 + (2 * r + 16 * lh)];
      bf = __builtin_bit_cast(v16bf, bu);
      s0 = __builtin_amdgcn_wmma_f32_16x16x32_bf16(false, aq1, false, bf, (short)0, s0, false, false);
#pragma unroll
      for (int r = 0; r < 8; ++r) bu[r] = *(const uint32_t*)&Kt[(16 + col) * KT_STRIDE + 0 + (2 * r + 16 * lh)];
      bf = __builtin_bit_cast(v16bf, bu);
      s1 = __builtin_amdgcn_wmma_f32_16x16x32_bf16(false, aq0, false, bf, (short)0, s1, false, false);
#pragma unroll
      for (int r = 0; r < 8; ++r) bu[r] = *(const uint32_t*)&Kt[(16 + col) * KT_STRIDE + 32 + (2 * r + 16 * lh)];
      bf = __builtin_bit_cast(v16bf, bu);
      s1 = __builtin_amdgcn_wmma_f32_16x16x32_bf16(false, aq1, false, bf, (short)0, s1, false, false);
    }

    // ---- bias + scale + online softmax (row = v + 8*lh, lives in one half) ----
    float p0[8], p1[8], corr[8];
#pragma unroll
    for (int v = 0; v < 8; ++v) {
      const int qrow = v + 8 * lh;
      const float rs = RSt[qrow * 32 + kt];          // row bias: scalar per tile
      const float c0 = CSt[qrow * 32 + col];         // col bias: same vec every tile
      const float c1 = CSt[qrow * 32 + col + 16];
      const float t0 = (s0[v] + rs + c0) * 0.125f;
      const float t1 = (s1[v] + rs + c1) * 0.125f;
      float mx = fmaxf(t0, t1);
      mx = fmaxf(mx, __shfl_xor(mx, 1, 32));
      mx = fmaxf(mx, __shfl_xor(mx, 2, 32));
      mx = fmaxf(mx, __shfl_xor(mx, 4, 32));
      mx = fmaxf(mx, __shfl_xor(mx, 8, 32));
      const float nm = fmaxf(rowm[v], mx);
      corr[v] = __expf(rowm[v] - nm);
      p0[v] = __expf(t0 - nm);
      p1[v] = __expf(t1 - nm);
      float sum = p0[v] + p1[v];
      sum += __shfl_xor(sum, 1, 32);
      sum += __shfl_xor(sum, 2, 32);
      sum += __shfl_xor(sum, 4, 32);
      sum += __shfl_xor(sum, 8, 32);
      rowl[v] = rowl[v] * corr[v] + sum;
      rowm[v] = nm;
    }
#pragma unroll
    for (int v = 0; v < 8; ++v) { o0[v] *= corr[v]; o1[v] *= corr[v]; o2[v] *= corr[v]; o3[v] *= corr[v]; }

    // ---- P: C-layout -> LDS -> A-layout ----
#pragma unroll
    for (int v = 0; v < 8; ++v) {
      const int qrow = v + 8 * lh;
      Pt[qrow * PT_STRIDE + col]      = f2bf_bits(p0[v]);
      Pt[qrow * PT_STRIDE + col + 16] = f2bf_bits(p1[v]);
    }
    __syncthreads();
    v8u pu;
#pragma unroll
    for (int r = 0; r < 8; ++r) {
      const int k = 2 * r + 16 * lh;
      pu[r] = *(const uint32_t*)&Pt[col * PT_STRIDE + k];
    }
    const v16bf pf = __builtin_bit_cast(v16bf, pu);

    // ---- O += P * V (four 16-wide d blocks) ----
    {
      v8u bu; v16bf bf;
#pragma unroll
      for (int r = 0; r < 8; ++r) bu[r] = *(const uint32_t*)&Vt[(0  + col) * VT_STRIDE + (2 * r + 16 * lh)];
      bf = __builtin_bit_cast(v16bf, bu);
      o0 = __builtin_amdgcn_wmma_f32_16x16x32_bf16(false, pf, false, bf, (short)0, o0, false, false);
#pragma unroll
      for (int r = 0; r < 8; ++r) bu[r] = *(const uint32_t*)&Vt[(16 + col) * VT_STRIDE + (2 * r + 16 * lh)];
      bf = __builtin_bit_cast(v16bf, bu);
      o1 = __builtin_amdgcn_wmma_f32_16x16x32_bf16(false, pf, false, bf, (short)0, o1, false, false);
#pragma unroll
      for (int r = 0; r < 8; ++r) bu[r] = *(const uint32_t*)&Vt[(32 + col) * VT_STRIDE + (2 * r + 16 * lh)];
      bf = __builtin_bit_cast(v16bf, bu);
      o2 = __builtin_amdgcn_wmma_f32_16x16x32_bf16(false, pf, false, bf, (short)0, o2, false, false);
#pragma unroll
      for (int r = 0; r < 8; ++r) bu[r] = *(const uint32_t*)&Vt[(48 + col) * VT_STRIDE + (2 * r + 16 * lh)];
      bf = __builtin_bit_cast(v16bf, bu);
      o3 = __builtin_amdgcn_wmma_f32_16x16x32_bf16(false, pf, false, bf, (short)0, o3, false, false);
    }
    __syncthreads();
  }

  // ---- epilogue: normalize and store mixed -> OB[b][n][h*64+d] ----
  float* ob = OB + ((size_t)b * 1024 + n0) * 512 + (size_t)(bh & 7) * 64;
#pragma unroll
  for (int v = 0; v < 8; ++v) {
    const int qrow = v + 8 * lh;
    const float inv = 1.0f / rowl[v];
    float* dst = ob + (size_t)qrow * 512;
    dst[0 * 16 + col] = o0[v] * inv;
    dst[1 * 16 + col] = o1[v] * inv;
    dst[2 * 16 + col] = o2[v] * inv;
    dst[3 * 16 + col] = o3[v] * inv;
  }
}

// ---------------------------------------------------------------------------
// Kernel 4: out = mixed @ Wv^T + bv   (fp32).  One block per token, 64 threads.
// ---------------------------------------------------------------------------
__global__ __launch_bounds__(64)
void vox_out(const float* __restrict__ OB, const float* __restrict__ Wv,
             const float* __restrict__ bv, float* __restrict__ out) {
  __shared__ float row[512];
  const int bn = blockIdx.x, t = threadIdx.x;
  const float* src = OB + (size_t)bn * 512;
  for (int i = t; i < 512; i += 64) row[i] = src[i];
  __syncthreads();
  const float* w = Wv + (size_t)t * 512;   // Wv is (64, 512) row-major
  float acc = bv[t];
#pragma unroll 8
  for (int e = 0; e < 512; ++e) acc += row[e] * w[e];
  out[(size_t)bn * 64 + t] = acc;
}

// ---------------------------------------------------------------------------
extern "C" void kernel_launch(void* const* d_in, const int* in_sizes, int n_in,
                              void* d_out, int out_size, void* d_ws, size_t ws_size,
                              hipStream_t stream) {
  (void)in_sizes; (void)n_in; (void)out_size; (void)ws_size;
  const float* hs      = (const float*)d_in[0];
  const float* row_emb = (const float*)d_in[1];
  const float* col_emb = (const float*)d_in[2];
  const float* Wq      = (const float*)d_in[3];
  const float* bq      = (const float*)d_in[4];
  const float* Wk      = (const float*)d_in[5];
  const float* bk      = (const float*)d_in[6];
  const float* Wv      = (const float*)d_in[7];
  const float* bv      = (const float*)d_in[8];
  float* out = (float*)d_out;

  char* w = (char*)d_ws;                       // needs >= 49 MB of scratch
  uint16_t* Qbf = (uint16_t*)(w);              //  8 MB
  uint16_t* Kbf = (uint16_t*)(w + (size_t)(8)  * (1u << 20));   //  8 MB
  uint16_t* Vbf = (uint16_t*)(w + (size_t)(16) * (1u << 20));   //  1 MB
  float*    RS  = (float*)   (w + (size_t)(17) * (1u << 20));   //  8 MB
  float*    CS  = (float*)   (w + (size_t)(25) * (1u << 20));   //  8 MB
  float*    OB  = (float*)   (w + (size_t)(33) * (1u << 20));   // 16 MB

  vox_proj<<<8192, 256, 0, stream>>>(hs, Wq, bq, Wk, bk, Qbf, Kbf, Vbf);
  vox_bias<<<8192, 256, 0, stream>>>(Qbf, row_emb, col_emb, RS, CS);
  dim3 grid_attn(64, 64);                      // (query tiles, b*NH)
  vox_attn<<<grid_attn, 32, 0, stream>>>(Qbf, Kbf, Vbf, RS, CS, OB);
  vox_out<<<8192, 64, 0, stream>>>(OB, Wv, bv, out);
}
